// LSTMHierarchialAttention3_3951369912415
// MI455X (gfx1250) — matrix operand order repaired and checked by reading, not compiled, not run on hardware
//
#include <hip/hip_runtime.h>
#include <hip/hip_bf16.h>
#include <math.h>

#define S_  32
#define W_  64
#define D_  256
#define H_  256
#define T_  (S_ * W_)     // 2048 word positions
#define G4_ (4 * H_)      // 1024 gates
#define NT_ (G4_ / 16)    // 64 N-tiles
#define KT_ (D_ / 32)     // 8 K-tiles

typedef __attribute__((ext_vector_type(16))) __bf16 v16bf;
typedef __attribute__((ext_vector_type(8)))  __bf16 v8bf;
typedef __attribute__((ext_vector_type(8)))  float  v8f;

__device__ __forceinline__ __bf16 f2bf(float x) {
  unsigned u = __builtin_bit_cast(unsigned, x);
  unsigned r = u + 0x7FFFu + ((u >> 16) & 1u);   // round-to-nearest-even
  unsigned short hs = (unsigned short)(r >> 16);
  return __builtin_bit_cast(__bf16, hs);
}

__device__ __forceinline__ float sigm(float x) { return 1.0f / (1.0f + expf(-x)); }

__device__ __forceinline__ v16bf cat8(v8bf lo, v8bf hi) {
  return __builtin_shufflevector(lo, hi, 0, 1, 2, 3, 4, 5, 6, 7,
                                         8, 9, 10, 11, 12, 13, 14, 15);
}

// async copy 16B/lane from global into LDS (ASYNCcnt-tracked, bypasses VGPRs)
__device__ __forceinline__ void async_g2l_b128(void* lds_dst, const void* gsrc) {
  unsigned ldsoff = (unsigned)(size_t)lds_dst;            // flat->LDS uses addr[31:0]
  unsigned long long ga = (unsigned long long)(size_t)gsrc;
  asm volatile("global_load_async_to_lds_b128 %0, %1, off"
               :: "v"(ldsoff), "v"(ga) : "memory");
}

// ---------------- cast X to bf16 ----------------
__global__ void k_cast_bf16(const float* __restrict__ src, __bf16* __restrict__ dst, int n) {
  int i = blockIdx.x * blockDim.x + threadIdx.x;
  if (i < n) dst[i] = f2bf(src[i]);
}

// ---------------- swizzle weight [N=G4_][K] f32 into WMMA-B fragment order (bf16) --------
// one thread per (tile, lane); each lane's 16 elements stored contiguously (32 B).
__global__ void __launch_bounds__(256) k_swz_w(const float* __restrict__ Wsrc,
                                               __bf16* __restrict__ dst, int Kdim) {
  const int g    = blockIdx.x * blockDim.x + threadIdx.x;
  const int lane = g & 31;
  const int tile = g >> 5;            // kt*NT_ + nt
  const int nt   = tile & (NT_ - 1);
  const int kt   = tile >> 6;
  if (kt >= Kdim / 32) return;
  const int n  = nt * 16 + (lane & 15);
  const int kb = kt * 32 + (lane >> 4) * 8;
  #pragma unroll
  for (int i = 0; i < 16; ++i)
    dst[(size_t)g * 16 + i] = f2bf(Wsrc[(size_t)n * Kdim + kb + i + (i >= 8 ? 8 : 0)]);
}

// ---------------- G1 = X * W_ih1^T + b1 via WMMA bf16 ----------------
__global__ void __launch_bounds__(256) k_gemm_g1(const __bf16* __restrict__ Xbf,
                                                 const __bf16* __restrict__ Wswz,
                                                 const float*  __restrict__ b1,
                                                 float* __restrict__ G1) {
  const unsigned lane = threadIdx.x & 31u;
  const unsigned wv   = threadIdx.x >> 5;     // 0..7, each wave owns 128 N-columns
  const unsigned m0   = blockIdx.x * 16u;
  const unsigned l15  = lane & 15u;
  const unsigned hi   = lane >> 4;            // 0 or 1

  v8f acc[8];
  #pragma unroll
  for (int nt = 0; nt < 8; ++nt)
    #pragma unroll
    for (int j = 0; j < 8; ++j) acc[nt][j] = 0.0f;

  const unsigned arow = m0 + l15;
  #pragma unroll
  for (int kt = 0; kt < KT_; ++kt) {          // K = 256 in 8 steps of 32
    const unsigned kb = (unsigned)kt * 32u + hi * 8u;
    const v8bf alo = *(const v8bf*)(Xbf + (size_t)arow * D_ + kb);
    const v8bf ahi = *(const v8bf*)(Xbf + (size_t)arow * D_ + kb + 16);
    const v16bf a  = cat8(alo, ahi);
    #pragma unroll
    for (int nt = 0; nt < 8; ++nt) {
      const unsigned tile = (unsigned)kt * NT_ + wv * 8u + (unsigned)nt;
      const v16bf b = *(const v16bf*)(Wswz + ((size_t)tile * 32u + lane) * 16u);
      acc[nt] = __builtin_amdgcn_wmma_f32_16x16x32_bf16(false, a, false, b,
                                                        (short)0, acc[nt], false, false);
    }
  }
  #pragma unroll
  for (int nt = 0; nt < 8; ++nt) {
    const unsigned n  = wv * 128u + (unsigned)nt * 16u + l15;
    const float    bb = b1[n];
    #pragma unroll
    for (int r = 0; r < 8; ++r) {
      const unsigned m = m0 + (unsigned)r + hi * 8u;
      G1[(size_t)m * G4_ + n] = acc[nt][r] + bb;
    }
  }
}

// ---------------- sequential word LSTM (2048 steps) ----------------
// recurrent matvec via WMMA with replicated-row A; B double-buffered across kt in
// registers; next step's G1 row async-prefetched into LDS (ASYNCcnt) during compute.
__global__ void __launch_bounds__(256) k_word_lstm(const float*  __restrict__ G1,    // [T_][G4_]
                                                   const __bf16* __restrict__ Wswz,  // fragment order
                                                   const float* __restrict__ h0,
                                                   const float* __restrict__ c0,
                                                   float* __restrict__ Hw) {         // [T_][H_]
  __shared__ float  cs[H_];
  __shared__ float  gs[G4_];
  __shared__ float  g1buf[2][G4_];
  __shared__ __bf16 hb[H_];
  const unsigned tid  = threadIdx.x;
  const unsigned lane = tid & 31u;
  const unsigned wv   = tid >> 5;
  const unsigned l15  = lane & 15u;
  const unsigned hi   = lane >> 4;

  cs[tid] = c0[tid];
  hb[tid] = f2bf(h0[tid]);
  // kick off async fetch of G1 row 0 (16 B per lane)
  async_g2l_b128(&g1buf[0][tid * 4], G1 + (size_t)0 * G4_ + tid * 4);
  __syncthreads();

  for (int t = 0; t < T_; ++t) {
    const int cb = t & 1;
    if (t + 1 < T_)   // stream next row while this step computes
      async_g2l_b128(&g1buf[cb ^ 1][tid * 4], G1 + (size_t)(t + 1) * G4_ + tid * 4);

    v8f acc[8];
    #pragma unroll
    for (int nt = 0; nt < 8; ++nt)
      #pragma unroll
      for (int j = 0; j < 8; ++j) acc[nt][j] = 0.0f;

    v16bf bb[2][8];
    #pragma unroll
    for (int nt = 0; nt < 8; ++nt)
      bb[0][nt] = *(const v16bf*)(Wswz + ((size_t)(wv * 8u + (unsigned)nt) * 32u + lane) * 16u);

    #pragma unroll
    for (int kt = 0; kt < KT_; ++kt) {
      const int cur = kt & 1, nxt = cur ^ 1;
      if (kt + 1 < KT_) {
        #pragma unroll
        for (int nt = 0; nt < 8; ++nt) {
          const unsigned tile = (unsigned)(kt + 1) * NT_ + wv * 8u + (unsigned)nt;
          bb[nxt][nt] = *(const v16bf*)(Wswz + ((size_t)tile * 32u + lane) * 16u);
        }
      }
      const unsigned kb = (unsigned)kt * 32u + hi * 8u;
      const v8bf alo = *(const v8bf*)(hb + kb);        // replicated-row h tile
      const v8bf ahi = *(const v8bf*)(hb + kb + 16);
      const v16bf a  = cat8(alo, ahi);
      #pragma unroll
      for (int nt = 0; nt < 8; ++nt)
        acc[nt] = __builtin_amdgcn_wmma_f32_16x16x32_bf16(false, a, false, bb[cur][nt],
                                                          (short)0, acc[nt], false, false);
    }

    if (lane < 16) {                          // row M=0 of each tile carries the matvec
      #pragma unroll
      for (int nt = 0; nt < 8; ++nt) {
        const unsigned n = wv * 128u + (unsigned)nt * 16u + lane;
        gs[n] = acc[nt][0];
      }
    }
    // ensure row t finished (in-order async: <=1 outstanding leaves only row t+1 in flight)
    if (t + 1 < T_) asm volatile("s_wait_asynccnt 0x1" ::: "memory");
    else            asm volatile("s_wait_asynccnt 0x0" ::: "memory");
    __syncthreads();
    {
      const float* g1r = g1buf[cb];
      float ig = sigm(gs[tid] + g1r[tid]);
      float fg = sigm(gs[H_ + tid] + g1r[H_ + tid]);
      float gg = tanhf(gs[2 * H_ + tid] + g1r[2 * H_ + tid]);
      float og = sigm(gs[3 * H_ + tid] + g1r[3 * H_ + tid]);
      float cn = fg * cs[tid] + ig * gg;
      float hn = og * tanhf(cn);
      cs[tid] = cn;
      hb[tid] = f2bf(hn);
      Hw[(size_t)t * H_ + tid] = hn;
    }
    __syncthreads();
  }
}

// ---------------- word attention scores: 537 MB stream of A1W, bandwidth bound ----------------
__global__ void __launch_bounds__(256) k_attn1(const float* __restrict__ Hw,
                                               const float* __restrict__ A1W,
                                               const float* __restrict__ A1B,
                                               const float* __restrict__ A1U,
                                               float* __restrict__ sc1) {
  __shared__ float hwv[H_];
  __shared__ float red[256];
  const int p = blockIdx.x, tid = threadIdx.x;
  hwv[tid] = Hw[(size_t)p * H_ + tid];
  __syncthreads();
  const float* Ap = A1W + (size_t)p * H_ * H_;
  float acc = 0.0f;
  #pragma unroll 8
  for (int h = 0; h < H_; ++h) {
    if ((h & 31) == 0 && h + 64 < H_)
      __builtin_prefetch(&Ap[(size_t)(h + 64) * H_ + tid], 0, 1);
    acc += hwv[h] * Ap[(size_t)h * H_ + tid];
  }
  float u = tanhf(acc + A1B[(size_t)p * H_ + tid]);
  red[tid] = u * A1U[(size_t)p * H_ + tid];
  __syncthreads();
  for (int s = 128; s > 0; s >>= 1) {
    if (tid < s) red[tid] += red[tid + s];
    __syncthreads();
  }
  if (tid == 0) sc1[p] = expf(red[0]);
}

// ---------------- softmax over words + weighted sentence vectors ----------------
__global__ void __launch_bounds__(256) k_sentvec(const float* __restrict__ Hw,
                                                 const float* __restrict__ sc1,
                                                 float* __restrict__ sentVec) {
  __shared__ float sw[W_];
  __shared__ float tot;
  const int s = blockIdx.x, tid = threadIdx.x;
  if (tid < W_) sw[tid] = sc1[s * W_ + tid];
  __syncthreads();
  if (tid == 0) { float t = 0.0f; for (int w = 0; w < W_; ++w) t += sw[w]; tot = t; }
  __syncthreads();
  float acc = 0.0f;
  for (int w = 0; w < W_; ++w)
    acc += Hw[(size_t)(s * W_ + w) * H_ + tid] * sw[w];
  sentVec[s * H_ + tid] = acc / tot;
}

// ---------------- sentence LSTM (32 steps, tiny) ----------------
__global__ void __launch_bounds__(256) k_sent_lstm(const float* __restrict__ sentVec,
                                                   const float* __restrict__ Wih2,
                                                   const float* __restrict__ Whh2,
                                                   const float* __restrict__ b2,
                                                   const float* __restrict__ h0,
                                                   const float* __restrict__ c0,
                                                   float* __restrict__ rowH) {
  __shared__ float xs[H_], hsd[H_], csd[H_], gs[G4_];
  const int tid = threadIdx.x;
  hsd[tid] = h0[tid];
  csd[tid] = c0[tid];
  __syncthreads();
  for (int t = 0; t < S_; ++t) {
    xs[tid] = sentVec[t * H_ + tid];
    __syncthreads();
    #pragma unroll
    for (int q = 0; q < 4; ++q) {
      const int row = q * H_ + tid;
      float acc = b2[row];
      for (int d = 0; d < H_; ++d)
        acc += Wih2[(size_t)row * H_ + d] * xs[d] + Whh2[(size_t)row * H_ + d] * hsd[d];
      gs[row] = acc;
    }
    __syncthreads();
    {
      float ig = sigm(gs[tid]);
      float fg = sigm(gs[H_ + tid]);
      float gg = tanhf(gs[2 * H_ + tid]);
      float og = sigm(gs[3 * H_ + tid]);
      float cn = fg * csd[tid] + ig * gg;
      float hn = og * tanhf(cn);
      csd[tid] = cn;
      hsd[tid] = hn;
      rowH[t * H_ + tid] = hn;
    }
    __syncthreads();
  }
}

// ---------------- sentence attention (stale query = Hw[last]) ----------------
__global__ void __launch_bounds__(256) k_attn2(const float* __restrict__ Hw,
                                               const float* __restrict__ A2W,
                                               const float* __restrict__ A2B,
                                               const float* __restrict__ A2U,
                                               float* __restrict__ sc2) {
  __shared__ float lastv[H_];
  __shared__ float red[256];
  const int s = blockIdx.x, tid = threadIdx.x;
  lastv[tid] = Hw[(size_t)(T_ - 1) * H_ + tid];
  __syncthreads();
  const float* Ap = A2W + (size_t)s * H_ * H_;
  float acc = 0.0f;
  #pragma unroll 8
  for (int h = 0; h < H_; ++h) {
    if ((h & 31) == 0 && h + 64 < H_)
      __builtin_prefetch(&Ap[(size_t)(h + 64) * H_ + tid], 0, 1);
    acc += lastv[h] * Ap[(size_t)h * H_ + tid];
  }
  float u = tanhf(acc + A2B[s * H_ + tid]);
  red[tid] = u * A2U[s * H_ + tid];
  __syncthreads();
  for (int st = 128; st > 0; st >>= 1) {
    if (tid < st) red[tid] += red[tid + st];
    __syncthreads();
  }
  if (tid == 0) sc2[s] = expf(red[0]);
}

// ---------------- final: softmax over sentences, weighted rowH, sigmoid(dot) ----------------
__global__ void __launch_bounds__(256) k_final(const float* __restrict__ rowH,
                                               const float* __restrict__ sc2,
                                               const float* __restrict__ Wf,
                                               const float* __restrict__ bfp,
                                               float* __restrict__ out) {
  __shared__ float a2[S_];
  __shared__ float red[256];
  __shared__ float tot;
  const int tid = threadIdx.x;
  if (tid < S_) a2[tid] = sc2[tid];
  __syncthreads();
  if (tid == 0) { float t = 0.0f; for (int s = 0; s < S_; ++s) t += a2[s]; tot = t; }
  __syncthreads();
  float o = 0.0f;
  for (int s = 0; s < S_; ++s) o += rowH[s * H_ + tid] * a2[s];
  o /= tot;
  red[tid] = o * Wf[tid];
  __syncthreads();
  for (int st = 128; st > 0; st >>= 1) {
    if (tid < st) red[tid] += red[tid + st];
    __syncthreads();
  }
  if (tid == 0) out[0] = 1.0f / (1.0f + expf(-(red[0] + bfp[0])));
}

extern "C" void kernel_launch(void* const* d_in, const int* in_sizes, int n_in,
                              void* d_out, int out_size, void* d_ws, size_t ws_size,
                              hipStream_t stream) {
  (void)in_sizes; (void)n_in; (void)out_size; (void)ws_size;
  const float* inputs = (const float*)d_in[0];
  const float* W_ih1  = (const float*)d_in[1];
  const float* W_hh1  = (const float*)d_in[2];
  const float* b1     = (const float*)d_in[3];
  const float* W_ih2  = (const float*)d_in[4];
  const float* W_hh2  = (const float*)d_in[5];
  const float* b2     = (const float*)d_in[6];
  const float* h1_0   = (const float*)d_in[7];
  const float* c1_0   = (const float*)d_in[8];
  const float* h2_0   = (const float*)d_in[9];
  const float* c2_0   = (const float*)d_in[10];
  const float* A1W    = (const float*)d_in[11];
  const float* A1B    = (const float*)d_in[12];
  const float* A1U    = (const float*)d_in[13];
  const float* A2W    = (const float*)d_in[14];
  const float* A2B    = (const float*)d_in[15];
  const float* A2U    = (const float*)d_in[16];
  const float* Wf     = (const float*)d_in[17];
  const float* bfp    = (const float*)d_in[18];
  float* out = (float*)d_out;

  // workspace carve-up (all 256B aligned)
  char* w = (char*)d_ws;
  size_t off = 0;
  auto carve = [&](size_t bytes) { char* p = w + off; off = (off + bytes + 255) & ~(size_t)255; return p; };
  __bf16* Xbf     = (__bf16*)carve((size_t)T_ * D_ * 2);    // 1 MB
  __bf16* WihSwz  = (__bf16*)carve((size_t)D_ * G4_ * 2);   // 0.5 MB, fragment order
  __bf16* WhhSwz  = (__bf16*)carve((size_t)H_ * G4_ * 2);   // 0.5 MB, fragment order
  float*  G1      = (float*) carve((size_t)T_ * G4_ * 4);   // 8 MB
  float*  Hw      = (float*) carve((size_t)T_ * H_ * 4);    // 2 MB
  float*  sc1     = (float*) carve((size_t)T_ * 4);
  float*  sentVec = (float*) carve((size_t)S_ * H_ * 4);
  float*  rowH    = (float*) carve((size_t)S_ * H_ * 4);
  float*  sc2     = (float*) carve((size_t)S_ * 4);

  // 1) bf16 cast of X; weights swizzled straight into WMMA-B fragment order
  k_cast_bf16<<<dim3((T_ * D_ + 255) / 256), dim3(256), 0, stream>>>(inputs, Xbf, T_ * D_);
  k_swz_w<<<dim3((KT_ * NT_ * 32) / 256), dim3(256), 0, stream>>>(W_ih1, WihSwz, D_);
  k_swz_w<<<dim3((KT_ * NT_ * 32) / 256), dim3(256), 0, stream>>>(W_hh1, WhhSwz, H_);

  // 2) hoisted input-gate GEMM (WMMA bf16): G1 = X * W_ih1^T + b1
  k_gemm_g1<<<dim3(T_ / 16), dim3(256), 0, stream>>>(Xbf, WihSwz, b1, G1);

  // 3) sequential word LSTM (WMMA recurrence + async-LDS G1 prefetch)
  k_word_lstm<<<dim3(1), dim3(256), 0, stream>>>(G1, WhhSwz, h1_0, c1_0, Hw);

  // 4) word attention scores (streams 537 MB of A1W; bandwidth-bound)
  k_attn1<<<dim3(T_), dim3(256), 0, stream>>>(Hw, A1W, A1B, A1U, sc1);

  // 5) per-sentence softmax + weighted sum -> sentence vectors
  k_sentvec<<<dim3(S_), dim3(256), 0, stream>>>(Hw, sc1, sentVec);

  // 6) sentence LSTM (32 steps)
  k_sent_lstm<<<dim3(1), dim3(256), 0, stream>>>(sentVec, W_ih2, W_hh2, b2, h2_0, c2_0, rowH);

  // 7) sentence attention with stale last-word query
  k_attn2<<<dim3(S_), dim3(256), 0, stream>>>(Hw, A2W, A2B, A2U, sc2);

  // 8) final reduction + sigmoid
  k_final<<<dim3(1), dim3(256), 0, stream>>>(rowH, sc2, Wf, bfp, out);
}
